// M_42365557408264
// MI455X (gfx1250) — matrix operand order repaired and checked by reading, not compiled, not run on hardware
//
#include <hip/hip_runtime.h>

// softmax(scores + bias) @ V  — fused single-pass (flash-style) kernel for
// B=8, H=12, S=1024, D=64, fp32 in/out, bf16 WMMA for the P·V matmul.
//
// Roofline: ~453 MB traffic (403 MB scores, single-use) -> ~19us at 23.3 TB/s.
// Scores stream with NT hints (V/bias stay L2-resident). Softmax runs in the
// log2 domain; online-max stats update once per 64-key stage. Row sums come
// from an extra WMMA against an all-ones B fragment (matrix pipe, not VALU),
// which also lands them in C-layout so the epilogue needs no broadcasts.

typedef __attribute__((ext_vector_type(16))) __bf16 v16bf;
typedef __attribute__((ext_vector_type(2)))  __bf16 bf16x2;
typedef __attribute__((ext_vector_type(8)))  float  v8f;
typedef __attribute__((ext_vector_type(4)))  float  f32x4;

#define SDIM   1024
#define DDIM   64
#define KSTAGE 64      // keys per LDS stage == softmax update granularity
#define WAVES  8
#define ROWS_PER_BLOCK (WAVES * 16)   // 128 q-rows per block

// cross-half (xor 16) reduction and row broadcast via LDS crossbar
static __device__ __forceinline__ float xor16_f(float x) {
  int r = __builtin_amdgcn_ds_swizzle(__builtin_bit_cast(int, x), 0x401F);
  return __builtin_bit_cast(float, r);
}
static __device__ __forceinline__ float bcast_lane_f(float x, int srcLane) {
  int r = __builtin_amdgcn_ds_bpermute(srcLane << 2, __builtin_bit_cast(int, x));
  return __builtin_bit_cast(float, r);
}

__launch_bounds__(256)
__global__ void attn_pv_kernel(const float* __restrict__ scores,
                               const float* __restrict__ v,
                               const float* __restrict__ bias,
                               float* __restrict__ out) {
  __shared__ __bf16 vlds[DDIM][KSTAGE];   // V stage, transposed [d][k], 8 KB
  __shared__ float  blds[SDIM];           // bias * log2(e), 4 KB

  const int tid  = threadIdx.x;
  const int wave = tid >> 5;
  const int lane = tid & 31;
  const int m    = lane & 15;   // row (A-frag) / column (B,C-frag) index
  const int g    = lane >> 4;   // lane half

  const int qtiles = SDIM / ROWS_PER_BLOCK;           // 8
  const int bh = blockIdx.x / qtiles;
  const int qt = blockIdx.x % qtiles;
  const int q0 = qt * ROWS_PER_BLOCK + wave * 16;     // this wave's 16 q-rows

  const float* srow  = scores + (size_t)bh * SDIM * SDIM + (size_t)(q0 + m) * SDIM;
  const float* vbase = v      + (size_t)bh * SDIM * DDIM;

  const float L2E = 1.44269504088896f;

  // one-time: bias pre-scaled into log2 domain (covered by first barrier)
  for (int i = tid; i < SDIM; i += 256) blds[i] = bias[i] * L2E;

  // all-ones bf16 B fragment for row-sum WMMA (lives in 8 VGPRs)
  v16bf bones;
#pragma unroll
  for (int i = 0; i < 16; ++i) bones[i] = (__bf16)1.0f;

  v8f acc[4] = {};            // 16x64 f32 output tile (4 N-tiles of 16)
  v8f lsum   = {};            // row sums in C layout (all columns identical)
  float mrun = -3.0e38f;      // running row max (log2 domain; lane&15 = row)

  for (int ks = 0; ks < SDIM; ks += KSTAGE) {
    // ---- cooperative V stage: global f32 [k][d] -> LDS bf16 [d][k] ----
    // each thread handles a k-pair so stores are packed b32 (cvt_pk path)
#pragma unroll
    for (int c = 0; c < 2; ++c) {
      int pi = tid + c * 256;            // pair index, 0..511
      int k0 = (pi >> 4) << 1;           // even k, 0..62
      int d0 = (pi & 15) << 2;           // 0..60
      const float* vp = vbase + (size_t)(ks + k0) * DDIM + d0;
      f32x4 v0 = *(const f32x4*)(vp);
      f32x4 v1 = *(const f32x4*)(vp + DDIM);
#pragma unroll
      for (int e = 0; e < 4; ++e) {
        bf16x2 pk = { (__bf16)v0[e], (__bf16)v1[e] };
        *(bf16x2*)&vlds[d0 + e][k0] = pk;
      }
    }
    __syncthreads();

    // ---- load 64 keys of scores (NT) + bias_l2e (LDS), fma to log2 domain --
    // A-frag K order per 32-chunk c: K = c*32 + 16g + 0..15
    const float* sp = srow + ks + 16 * g;
    const float* bp = blds + ks + 16 * g;
    float tv[32];
#pragma unroll
    for (int j = 0; j < 8; ++j) {
      const int off = (j >> 2) * 32 + (j & 3) * 4;   // 0,4,8,12,32,36,40,44
      f32x4 sv = __builtin_nontemporal_load((const f32x4*)(sp + off));
      f32x4 bv = *(const f32x4*)(bp + off);
#pragma unroll
      for (int e = 0; e < 4; ++e) tv[j * 4 + e] = sv[e] * L2E + bv[e];
    }

    // ---- online max, log-depth tree (ILP), one update per 64-key stage ----
    float mx[16];
#pragma unroll
    for (int i = 0; i < 16; ++i) mx[i] = fmaxf(tv[2 * i], tv[2 * i + 1]);
#pragma unroll
    for (int s2 = 8; s2 >= 1; s2 >>= 1) {
#pragma unroll
      for (int i = 0; i < s2; ++i) mx[i] = fmaxf(mx[i], mx[i + s2]);
    }
    float smax = fmaxf(mx[0], xor16_f(mx[0]));
    float mnew = fmaxf(mrun, smax);
    float corr = __builtin_amdgcn_exp2f(mrun - mnew);
    mrun = mnew;

    // ---- bulk subtract, then bulk exp2 (independent trans ops pipeline) ----
#pragma unroll
    for (int i = 0; i < 32; ++i) tv[i] -= mnew;
#pragma unroll
    for (int i = 0; i < 32; ++i) tv[i] = __builtin_amdgcn_exp2f(tv[i]);

    // ---- convert both chunks' A fragments up front (no WAR on WMMA) ----
    v16bf afrag[2];
#pragma unroll
    for (int c = 0; c < 2; ++c)
#pragma unroll
      for (int i = 0; i < 16; ++i) afrag[c][i] = (__bf16)tv[c * 16 + i];

    // ---- rescale accumulators: C-layout row for VGPR r is (8g + r) ----
#pragma unroll
    for (int r = 0; r < 8; ++r) {
      float sr = bcast_lane_f(corr, g * 8 + r);
      acc[0][r] *= sr;
      acc[1][r] *= sr;
      acc[2][r] *= sr;
      acc[3][r] *= sr;
      lsum[r]   *= sr;
    }

    // ---- WMMA: 4 D-tiles + 1 row-sum per 32-key chunk ----
#pragma unroll
    for (int c = 0; c < 2; ++c) {
      lsum = __builtin_amdgcn_wmma_f32_16x16x32_bf16(
          false, afrag[c], false, bones, (short)0, lsum, false, false);
#pragma unroll
      for (int t = 0; t < 4; ++t) {
        // B 32x16 bf16: lane half g holds K = 16g..16g+15, column n = lane&15
        v16bf bfrag = *(const v16bf*)&vlds[t * 16 + m][c * 32 + 16 * g];
        acc[t] = __builtin_amdgcn_wmma_f32_16x16x32_bf16(
            false, afrag[c], false, bfrag, (short)0, acc[t], false, false);
      }
    }
    __syncthreads();
  }

  // ---- epilogue: normalize by 1/rowsum (already in C layout), NT store ----
  float* obase = out + (size_t)bh * SDIM * DDIM;
#pragma unroll
  for (int r = 0; r < 8; ++r) {
    float linv = __builtin_amdgcn_rcpf(lsum[r]);
    int row = q0 + g * 8 + r;
#pragma unroll
    for (int t = 0; t < 4; ++t) {
      __builtin_nontemporal_store(acc[t][r] * linv,
                                  obase + (size_t)row * DDIM + t * 16 + m);
    }
  }
}

extern "C" void kernel_launch(void* const* d_in, const int* in_sizes, int n_in,
                              void* d_out, int out_size, void* d_ws, size_t ws_size,
                              hipStream_t stream) {
  const float* scores = (const float*)d_in[0];
  const float* v      = (const float*)d_in[1];
  const float* bias   = (const float*)d_in[2];
  float* out          = (float*)d_out;

  const int bhTotal = in_sizes[0] / (SDIM * SDIM);           // B*H = 96
  const int blocks  = bhTotal * (SDIM / ROWS_PER_BLOCK);     // 96*8 = 768

  attn_pv_kernel<<<blocks, 256, 0, stream>>>(scores, v, bias, out);
}
